// FastAttention_57286273794645
// MI455X (gfx1250) — matrix operand order repaired and checked by reading, not compiled, not run on hardware
//
#include <hip/hip_runtime.h>
#include <hip/hip_bf16.h>

#define HH 4
#define NN 2048
#define DD 64
#define MM 256
#define EE 64
#define TT 128
#define CC (NN/TT)   // 16 chunks per head

typedef __attribute__((ext_vector_type(16))) __bf16 v16bf;
typedef __attribute__((ext_vector_type(8)))  __bf16 v8bf;
typedef __attribute__((ext_vector_type(8)))  float  v8f;

union bffrag { v16bf v; v8bf h[2]; };

// ---- CDNA5 WMMA fragment index maps (wave32) ----
// A 16x32 bf16: lane row = lane&15; kb=(lane&16)?8:0; elem i<8 -> K=kb+i ; else K=16+kb+(i-8)
__device__ __forceinline__ int fkA(int lane, int i) {
    int kb = (lane & 16) ? 8 : 0;
    return (i < 8) ? (kb + i) : (16 + kb + (i - 8));
}
// B 32x16 bf16: lane col = lane&15; kb=(lane&16)?8:0; elem i<8 -> K=kb+i ; else K=16+kb+(i-8)
__device__ __forceinline__ int fkB(int lane, int i) {
    int kb = (lane & 16) ? 8 : 0;
    return (i < 8) ? (kb + i) : (16 + kb + (i - 8));
}
__device__ __forceinline__ v8f wmma_bf16(v16bf a, v16bf b, v8f c) {
    return __builtin_amdgcn_wmma_f32_16x16x32_bf16(false, a, false, b, (short)0, c, false, false);
}
__device__ __forceinline__ v8f zero8() {
    v8f z = {0.f,0.f,0.f,0.f,0.f,0.f,0.f,0.f}; return z;
}
// A-fragment from row-major bf16 (two contiguous 8-element runs per lane)
__device__ __forceinline__ v16bf loadA_bf16(const __bf16* base, int ld, int row, int kbase, int lane) {
    const __bf16* p = base + (size_t)row * ld + kbase + ((lane & 16) ? 8 : 0);
    bffrag f;
    f.h[0] = *(const v8bf*)p;
    f.h[1] = *(const v8bf*)(p + 16);
    return f.v;
}
// B-fragment from K-contiguous column (two 8-element runs per lane: kb and 16+kb, kb=8 for upper lane-half)
__device__ __forceinline__ v16bf loadB_run(const __bf16* p0, int lane) {
    const __bf16* p = p0 + ((lane & 16) ? 8 : 0);
    bffrag f;
    f.h[0] = *(const v8bf*)p;
    f.h[1] = *(const v8bf*)(p + 16);
    return f.v;
}
// ordered-uint encoding so atomicMax on uint == float max
__device__ __forceinline__ unsigned fkey(float x) {
    unsigned u = __float_as_uint(x);
    return (u & 0x80000000u) ? ~u : (u | 0x80000000u);
}
__device__ __forceinline__ float funkey(unsigned k) {
    unsigned u = (k & 0x80000000u) ? (k & 0x7FFFFFFFu) : ~k;
    return __uint_as_float(u);
}

#define CN    0.35355339059327379f   // 64^-0.25
#define CN2H  (0.5f * CN * CN)
#define RATIO 0.0625f                // 256^-0.5
#define FEPS  1e-4f

// ---------------- init ----------------
__global__ void init_kernel(unsigned* gmax) { *gmax = 0u; }

// ---------------- pack P^T into fragment-ordered bf16 ----------------
__global__ void pack_P(const float* __restrict__ P, __bf16* __restrict__ Pf) {
    int idx = blockIdx.x * blockDim.x + threadIdx.x;
    if (idx >= 16 * 2 * 32) return;
    int lane = idx & 31;
    int ks = (idx >> 5) & 1;
    int mt = idx >> 6;
    int col = mt * 16 + (lane & 15);
    #pragma unroll
    for (int i = 0; i < 16; ++i)
        Pf[idx * 16 + i] = (__bf16)P[col * DD + ks * 32 + fkB(lane, i)];
}

// ---------------- v -> v^T bf16 ([H][E][N]) via LDS tiles ----------------
__global__ __launch_bounds__(256) void pack_v_t(const float* __restrict__ v,
                                                __bf16* __restrict__ vt) {
    __shared__ __bf16 tile[64][65];
    const int bid = blockIdx.x;            // h*(NN/64) + nt
    const int h = bid >> 5, nt = bid & 31;
    const float* src = v + (size_t)h * NN * EE + (size_t)nt * 64 * EE;
    __bf16* dst = vt + (size_t)h * EE * NN + nt * 64;
    const int t = threadIdx.x;
    for (int idx = t; idx < 64 * 64; idx += 256) {
        int r = idx >> 6, c = idx & 63;
        tile[r][c] = (__bf16)src[r * EE + c];
    }
    __syncthreads();
    for (int idx = t; idx < 64 * 64; idx += 256) {
        int r = idx >> 6, c = idx & 63;
        dst[(size_t)r * NN + c] = tile[c][r];
    }
}

// ---------------- k_feat -> k_feat^T bf16 ([H][M][N]) via LDS tiles ----------------
__global__ __launch_bounds__(256) void transpose_kf(const __bf16* __restrict__ kf,
                                                    __bf16* __restrict__ kft) {
    __shared__ __bf16 tile[64][65];
    const int bid = blockIdx.x;            // h*128 + nt*4 + mt
    const int h = bid >> 7, rem = bid & 127;
    const int nt = rem >> 2, mt = rem & 3;
    const __bf16* src = kf + (size_t)h * NN * MM;
    __bf16* dst = kft + (size_t)h * MM * NN;
    const int t = threadIdx.x;
    for (int idx = t; idx < 64 * 64; idx += 256) {
        int r = idx >> 6, c = idx & 63;
        tile[r][c] = src[(size_t)(nt * 64 + r) * MM + mt * 64 + c];
    }
    __syncthreads();
    for (int idx = t; idx < 64 * 64; idx += 256) {
        int r = idx >> 6, c = idx & 63;
        dst[(size_t)(mt * 64 + r) * NN + nt * 64 + c] = tile[c][r];
    }
}

// ---------------- q features: 4 waves/block, 4 m-tiles/wave ----------------
__global__ __launch_bounds__(128) void feat_q_kernel(const float* __restrict__ q,
                                                     const __bf16* __restrict__ Pf,
                                                     __bf16* __restrict__ qf) {
    __shared__ float smax[4][16];
    const int t = threadIdx.x, lane = t & 31, w = t >> 5;
    const int h = blockIdx.x >> 7, rt = blockIdx.x & 127, row0 = rt * 16;
    const float* qh = q + (size_t)h * NN * DD;

    v16bf afr[2];
    const int arow = row0 + (lane & 15);
    #pragma unroll
    for (int ks = 0; ks < 2; ++ks)
        #pragma unroll
        for (int i = 0; i < 16; ++i)
            afr[ks][i] = (__bf16)(CN * qh[arow * DD + ks * 32 + fkA(lane, i)]);

    v8f acc[4];
    #pragma unroll
    for (int mi = 0; mi < 4; ++mi) {
        const int mt = w * 4 + mi;
        acc[mi] = zero8();
        #pragma unroll
        for (int ks = 0; ks < 2; ++ks) {
            v16bf bfr = *(const v16bf*)(Pf + (size_t)((mt * 2 + ks) * 32 + lane) * 16);
            acc[mi] = wmma_bf16(afr[ks], bfr, acc[mi]);
        }
    }

    float rm[8];
    #pragma unroll
    for (int r = 0; r < 8; ++r) {
        float vmx = acc[0][r];
        #pragma unroll
        for (int mi = 1; mi < 4; ++mi) vmx = fmaxf(vmx, acc[mi][r]);
        rm[r] = vmx;
    }
    #pragma unroll
    for (int r = 0; r < 8; ++r)
        for (int m = 1; m < 16; m <<= 1)
            rm[r] = fmaxf(rm[r], __shfl_xor(rm[r], m, 32));
    if ((lane & 15) == 0) {
        #pragma unroll
        for (int r = 0; r < 8; ++r) smax[w][((lane & 16) >> 1) + r] = rm[r];
    }
    __syncthreads();
    float rmax[8];
    #pragma unroll
    for (int r = 0; r < 8; ++r) {
        int rr = ((lane & 16) >> 1) + r;
        rmax[r] = fmaxf(fmaxf(smax[0][rr], smax[1][rr]), fmaxf(smax[2][rr], smax[3][rr]));
    }

    float ss = 0.f;
    #pragma unroll 8
    for (int d = 0; d < DD; ++d) { float x = qh[arow * DD + d]; ss += x * x; }
    float dg[8];
    #pragma unroll
    for (int r = 0; r < 8; ++r) {
        int rr = ((lane & 16) >> 1) + r;
        dg[r] = __shfl(ss, (lane & 16) | rr, 32) * CN2H;
    }

    __bf16* qfh = qf + (size_t)h * NN * MM;
    #pragma unroll
    for (int mi = 0; mi < 4; ++mi) {
        const int col = (w * 4 + mi) * 16 + (lane & 15);
        #pragma unroll
        for (int r = 0; r < 8; ++r) {
            int rowg = row0 + ((lane & 16) >> 1) + r;
            qfh[(size_t)rowg * MM + col] =
                (__bf16)(RATIO * (expf(acc[mi][r] - dg[r] - rmax[r]) + FEPS));
        }
    }
}

// ---------------- k features pass 1: dash(f32), diag, global max ----------------
__global__ __launch_bounds__(128) void feat_k_kernel(const float* __restrict__ k,
                                                     const __bf16* __restrict__ Pf,
                                                     float* __restrict__ kdash,
                                                     float* __restrict__ kdiag,
                                                     unsigned* __restrict__ gmax) {
    __shared__ float swm[4];
    const int t = threadIdx.x, lane = t & 31, w = t >> 5;
    const int h = blockIdx.x >> 7, rt = blockIdx.x & 127, row0 = rt * 16;
    const float* kh = k + (size_t)h * NN * DD;

    v16bf afr[2];
    const int arow = row0 + (lane & 15);
    #pragma unroll
    for (int ks = 0; ks < 2; ++ks)
        #pragma unroll
        for (int i = 0; i < 16; ++i)
            afr[ks][i] = (__bf16)(CN * kh[arow * DD + ks * 32 + fkA(lane, i)]);

    v8f acc[4];
    #pragma unroll
    for (int mi = 0; mi < 4; ++mi) {
        const int mt = w * 4 + mi;
        acc[mi] = zero8();
        #pragma unroll
        for (int ks = 0; ks < 2; ++ks) {
            v16bf bfr = *(const v16bf*)(Pf + (size_t)((mt * 2 + ks) * 32 + lane) * 16);
            acc[mi] = wmma_bf16(afr[ks], bfr, acc[mi]);
        }
    }

    float* kdh = kdash + (size_t)h * NN * MM;
    float wm = acc[0][0];
    #pragma unroll
    for (int mi = 0; mi < 4; ++mi) {
        const int col = (w * 4 + mi) * 16 + (lane & 15);
        #pragma unroll
        for (int r = 0; r < 8; ++r) {
            int rowg = row0 + ((lane & 16) >> 1) + r;
            kdh[(size_t)rowg * MM + col] = acc[mi][r];
            wm = fmaxf(wm, acc[mi][r]);
        }
    }
    for (int m = 1; m < 32; m <<= 1) wm = fmaxf(wm, __shfl_xor(wm, m, 32));
    if (lane == 0) swm[w] = wm;
    __syncthreads();
    if (t == 0)
        atomicMax(gmax, fkey(fmaxf(fmaxf(swm[0], swm[1]), fmaxf(swm[2], swm[3]))));

    if (w == 0) {
        float ss = 0.f;
        #pragma unroll 8
        for (int d = 0; d < DD; ++d) { float x = kh[arow * DD + d]; ss += x * x; }
        if (lane < 16) kdiag[(size_t)h * NN + row0 + lane] = ss * CN2H;
    }
}

// ---------------- k features pass 2: exp -> bf16 ----------------
__global__ void kfeat_finalize(const float* __restrict__ kdash, const float* __restrict__ kdiag,
                               const unsigned* __restrict__ gmax, __bf16* __restrict__ kf) {
    const float g = funkey(*gmax);
    const size_t total = (size_t)HH * NN * MM;
    for (size_t i = (size_t)blockIdx.x * blockDim.x + threadIdx.x; i < total;
         i += (size_t)gridDim.x * blockDim.x) {
        size_t n = i / MM;
        kf[i] = (__bf16)(RATIO * (expf(kdash[i] - kdiag[n] - g) + FEPS));
    }
}

// ---------------- per-chunk sums: Ssum = K^T V (f32 out), ksum = colsum(K) ----------------
__global__ __launch_bounds__(256) void chunk_sums(const __bf16* __restrict__ kf,
                                                  const __bf16* __restrict__ kft,
                                                  const __bf16* __restrict__ vt,
                                                  float* __restrict__ ssum,
                                                  float* __restrict__ ksum) {
    const int h = blockIdx.x >> 4, c = blockIdx.x & 15;
    const int n0 = c * TT;
    const __bf16* kfh  = kf  + (size_t)h * NN * MM;
    const __bf16* kfth = kft + (size_t)h * MM * NN;
    const __bf16* vth  = vt  + (size_t)h * EE * NN;
    const int t = threadIdx.x;

    {   // column sums of K-chunk (coalesced across threads)
        float s = 0.f;
        for (int j = 0; j < TT; ++j) s += (float)kfh[(size_t)(n0 + j) * MM + t];
        ksum[((size_t)h * CC + c) * MM + t] = s;
    }

    const int lane = t & 31, w = t >> 5;   // 8 waves x 8 tiles = 64 (m,e) tiles
    float* out = ssum + ((size_t)h * CC + c) * MM * EE;
    for (int qq = 0; qq < 8; ++qq) {
        const int id = w * 8 + qq;
        const int mtile = id >> 2, etile = id & 3;
        v8f acc = zero8();
        const int mrow = mtile * 16 + (lane & 15);
        const int ecol = etile * 16 + (lane & 15);
        #pragma unroll
        for (int ks = 0; ks < 4; ++ks) {
            v16bf a = loadA_bf16(kfth, NN, mrow, n0 + ks * 32, lane);
            v16bf b = loadB_run(vth + (size_t)ecol * NN + n0 + ks * 32, lane);
            acc = wmma_bf16(a, b, acc);
        }
        #pragma unroll
        for (int r = 0; r < 8; ++r) {
            int m = mtile * 16 + ((lane & 16) >> 1) + r;
            out[(size_t)m * EE + ecol] = acc[r];
        }
    }
}

// ---------------- exclusive prefix over chunks; S_prefix stored transposed bf16 [H][C][E][M] ----------------
__global__ void chunk_prefix(const float* __restrict__ ssum, const float* __restrict__ ksum,
                             __bf16* __restrict__ spret, float* __restrict__ kpre) {
    const int h = blockIdx.x, t = threadIdx.x;   // 256 threads, t == m
    {
        float run = 0.f;
        for (int c = 0; c < CC; ++c) {
            size_t idx = ((size_t)h * CC + c) * MM + t;
            kpre[idx] = run; run += ksum[idx];
        }
    }
    for (int ee = 0; ee < EE; ++ee) {
        float run = 0.f;
        for (int c = 0; c < CC; ++c) {
            spret[(((size_t)h * CC + c) * EE + ee) * MM + t] = (__bf16)run;  // coalesced
            run += ssum[((size_t)h * CC + c) * (MM * EE) + (size_t)t * EE + ee];
        }
    }
}

// ---------------- per-chunk output ----------------
#define SQP 264   // padded bf16 row stride for Q'
#define SAP 136   // padded bf16 row stride for intra A
__global__ __launch_bounds__(256) void chunk_out(const __bf16* __restrict__ qf,
                                                 const __bf16* __restrict__ kf,
                                                 const __bf16* __restrict__ vt,
                                                 const __bf16* __restrict__ spret,
                                                 const float* __restrict__ kpre,
                                                 float* __restrict__ out) {
    __shared__ __bf16 sq[TT * SQP];   // Q' = qf / Kc   (128 x 256, padded)
    __shared__ __bf16 sA[TT * SAP];   // tril(Q' K^T)   (128 x 128, padded)
    const int h = blockIdx.x >> 4, c = blockIdx.x & 15;
    const int n0 = c * TT;
    const __bf16* qfh = qf + (size_t)h * NN * MM;
    const __bf16* kfh = kf + (size_t)h * NN * MM;
    const __bf16* vth = vt + (size_t)h * EE * NN;
    const int t = threadIdx.x;

    // 1) running denominator Kc and Q' (one column per thread)
    {
        float kc = kpre[((size_t)h * CC + c) * MM + t];
        for (int j = 0; j < TT; ++j) {
            kc += (float)kfh[(size_t)(n0 + j) * MM + t];
            sq[j * SQP + t] = (__bf16)((float)qfh[(size_t)(n0 + j) * MM + t] / kc);
        }
    }
    __syncthreads();

    const int lane = t & 31, w = t >> 5;

    // 2) A = tril(Q' @ K_chunk^T)  -> bf16 in LDS
    for (int qq = 0; qq < 8; ++qq) {
        const int id = w * 8 + qq;
        const int jt = id >> 3, it = id & 7;
        const int col = it * 16 + (lane & 15);
        if (it > jt) {
            #pragma unroll
            for (int r = 0; r < 8; ++r) {
                int row = jt * 16 + ((lane & 16) >> 1) + r;
                sA[row * SAP + col] = (__bf16)0.f;
            }
            continue;
        }
        v8f acc = zero8();
        const int arow = jt * 16 + (lane & 15);
        const __bf16* bbase = kfh + (size_t)(n0 + col) * MM;   // m-contiguous row; two 8-runs of m per lane
        #pragma unroll
        for (int ks = 0; ks < 8; ++ks) {
            v16bf a = loadA_bf16(sq, SQP, arow, ks * 32, lane);
            v16bf b = loadB_run(bbase + ks * 32, lane);
            acc = wmma_bf16(a, b, acc);
        }
        #pragma unroll
        for (int r = 0; r < 8; ++r) {
            int row = jt * 16 + ((lane & 16) >> 1) + r;
            sA[row * SAP + col] = (__bf16)((col <= row) ? acc[r] : 0.f);
        }
    }
    __syncthreads();

    // 3) out = Q' @ S_prefix + A @ V_chunk
    const __bf16* spt = spret + ((size_t)h * CC + c) * EE * MM;   // [E][M]
    float* oh = out + (size_t)h * NN * EE;
    for (int qq = 0; qq < 4; ++qq) {
        const int id = w * 4 + qq;
        const int jt = id >> 2, et = id & 3;
        const int arow = jt * 16 + (lane & 15);
        const int ecol = et * 16 + (lane & 15);
        v8f acc = zero8();
        #pragma unroll
        for (int ks = 0; ks < 8; ++ks) {     // inter-chunk: K-dim = M = 256
            v16bf a = loadA_bf16(sq, SQP, arow, ks * 32, lane);
            v16bf b = loadB_run(spt + (size_t)ecol * MM + ks * 32, lane);
            acc = wmma_bf16(a, b, acc);
        }
        #pragma unroll
        for (int ks = 0; ks < 4; ++ks) {     // intra-chunk: K-dim = T = 128
            v16bf a = loadA_bf16(sA, SAP, arow, ks * 32, lane);
            v16bf b = loadB_run(vth + (size_t)ecol * NN + n0 + ks * 32, lane);
            acc = wmma_bf16(a, b, acc);
        }
        #pragma unroll
        for (int r = 0; r < 8; ++r) {
            int row = n0 + jt * 16 + ((lane & 16) >> 1) + r;
            oh[(size_t)row * EE + ecol] = acc[r];
        }
    }
}

extern "C" void kernel_launch(void* const* d_in, const int* in_sizes, int n_in,
                              void* d_out, int out_size, void* d_ws, size_t ws_size,
                              hipStream_t stream) {
    const float* q = (const float*)d_in[0];
    const float* k = (const float*)d_in[1];
    const float* v = (const float*)d_in[2];
    const float* P = (const float*)d_in[3];
    float* out = (float*)d_out;

    // ---- workspace layout (f32 first, then bf16; all >=64B aligned) ----
    float* kdash = (float*)d_ws;                           // [H,N,M]   f32
    float* ssum  = kdash + (size_t)HH * NN * MM;           // [H,C,M,E] f32
    float* kdiag = ssum  + (size_t)HH * CC * MM * EE;      // [H,N]     f32
    float* kpre  = kdiag + (size_t)HH * NN;                // [H,C,M]   f32
    float* ksum  = kpre  + (size_t)HH * CC * MM;           // [H,C,M]   f32
    unsigned* gmax = (unsigned*)(ksum + (size_t)HH * CC * MM);
    __bf16* qf    = (__bf16*)(gmax + 16);                  // [H,N,M]   bf16
    __bf16* kf    = qf  + (size_t)HH * NN * MM;            // [H,N,M]   bf16
    __bf16* kft   = kf  + (size_t)HH * NN * MM;            // [H,M,N]   bf16 (k_feat^T)
    __bf16* vt    = kft + (size_t)HH * MM * NN;            // [H,E,N]   bf16 (v^T)
    __bf16* spret = vt  + (size_t)HH * EE * NN;            // [H,C,E,M] bf16 (S_prefix^T)
    __bf16* Pf    = spret + (size_t)HH * CC * EE * MM;     // packed P fragments

    init_kernel<<<1, 1, 0, stream>>>(gmax);
    pack_P<<<4, 256, 0, stream>>>(P, Pf);
    pack_v_t<<<HH * (NN / 64), 256, 0, stream>>>(v, vt);
    feat_q_kernel<<<HH * (NN / 16), 128, 0, stream>>>(q, Pf, qf);
    feat_k_kernel<<<HH * (NN / 16), 128, 0, stream>>>(k, Pf, kdash, kdiag, gmax);
    kfeat_finalize<<<512, 256, 0, stream>>>(kdash, kdiag, gmax, kf);
    transpose_kf<<<HH * (NN / 64) * (MM / 64), 256, 0, stream>>>(kf, kft);
    chunk_sums<<<HH * CC, 256, 0, stream>>>(kf, kft, vt, ssum, ksum);
    chunk_prefix<<<HH, 256, 0, stream>>>(ssum, ksum, spret, kpre);
    chunk_out<<<HH * CC, 256, 0, stream>>>(qf, kf, vt, spret, kpre, out);
}
